// HSRnnBase_6141803233964
// MI455X (gfx1250) — matrix operand order repaired and checked by reading, not compiled, not run on hardware
//
#include <hip/hip_runtime.h>
#include <cstdint>

// ---------------------------------------------------------------------------
// Types for CDNA5 WMMA
// ---------------------------------------------------------------------------
typedef __attribute__((ext_vector_type(16))) _Float16 v16h;
typedef __attribute__((ext_vector_type(8)))  _Float16 v8h;
typedef __attribute__((ext_vector_type(8)))  float    v8f;

#define ALPHA 0.9f
#define BETA  0.9f
#define THETA 1.0f

// Problem sizes (fixed by the reference)
#define BDIM 64
#define TDIM 512
#define DDIM 512
#define H1   1024
#define H2   1024

// ---------------------------------------------------------------------------
// One-time prep kernels
// ---------------------------------------------------------------------------
__global__ __launch_bounds__(256) void f32_to_f16_kernel(
    const float* __restrict__ in, _Float16* __restrict__ out, int n) {
  int i = blockIdx.x * 256 + threadIdx.x;
  if (i < n) out[i] = (_Float16)in[i];
}

// Pack [W_in (K0 x N); W_rec (K1 x N)] into transposed f16 Wt (N x (K0+K1)),
// so WMMA B-fragments are two contiguous 8-half chunks per lane.
__global__ __launch_bounds__(256) void pack_wt_kernel(
    const float* __restrict__ Win, const float* __restrict__ Wrec,
    _Float16* __restrict__ Wt, int K0, int K1, int N) {
  int idx = blockIdx.x * 256 + threadIdx.x;
  int Ktot = K0 + K1;
  if (idx >= N * Ktot) return;
  int n = idx / Ktot;
  int k = idx - n * Ktot;
  float w = (k < K0) ? Win[(size_t)k * N + n] : Wrec[(size_t)(k - K0) * N + n];
  Wt[idx] = (_Float16)w;
}

// ---------------------------------------------------------------------------
// WMMA fragment loaders (contiguous b128 pairs per the ISA VGPR layouts)
// ---------------------------------------------------------------------------
// A (16-bit 16x32): lanes 0-15 hold K=0..7 & 16..23; lanes 16-31 K=8..15 & 24..31
__device__ __forceinline__ v16h load_a_frag(const _Float16* __restrict__ ar,
                                            int kb, int hh) {
  v16h a;
  ((v8h*)&a)[0] = *(const v8h*)(ar + kb + 8 * hh);
  ((v8h*)&a)[1] = *(const v8h*)(ar + kb + 16 + 8 * hh);
  return a;
}
// B (32x16): per lane, K = 8*hh + 0..7 & 16 + 8*hh + 0..7 down column n (contiguous in N-major Wt)
__device__ __forceinline__ v16h load_b_frag(const _Float16* __restrict__ wrow,
                                            int kb, int hh) {
  v16h b;
  ((v8h*)&b)[0] = *(const v8h*)(wrow + kb + 8 * hh);
  ((v8h*)&b)[1] = *(const v8h*)(wrow + kb + 16 + 8 * hh);
  return b;
}

#define WMMA_F16(A, B, C) \
  __builtin_amdgcn_wmma_f32_16x16x32_f16(false, (A), false, (B), (short)0, (C), false, false)

// ---------------------------------------------------------------------------
// One LIF layer timestep:
//   pre  = [Ax | As] @ [W_in; W_rec]^T-packed  (WMMA f16 -> f32)
//   c    = ALPHA*c + pre + bias
//   v    = BETA*v + c ; sp = (v-THETA > 0) ; v -= sp*THETA
// Grid: N/32 blocks of 128 threads (4 waves). Wave w owns M-tile w (B=64);
// each wave produces TWO adjacent 16x16 N-tiles (A reuse) with TWO independent
// accumulator chains per tile (breaks the WMMA RAW dependency chain).
// ---------------------------------------------------------------------------
__global__ __launch_bounds__(128) void lif_step_kernel(
    const _Float16* __restrict__ Ax, int sAx, int K0,   // input segment (B rows, stride sAx)
    const _Float16* __restrict__ As, int K1,            // recurrent spikes (B x K1, dense)
    const _Float16* __restrict__ Wt,                    // N x (K0+K1), transposed packed
    const float* __restrict__ bias,
    float* __restrict__ c, float* __restrict__ v,
    _Float16* __restrict__ s_out,                       // B x N f16 spikes out
    float* __restrict__ y_out,                          // optional f32 spikes (final step)
    int N) {
  const int lane = threadIdx.x & 31;
  const int wave = threadIdx.x >> 5;   // m-tile 0..3
  const int m    = lane & 15;
  const int hh   = lane >> 4;          // half-wave select
  const int Ktot = K0 + K1;
  const int nt0  = blockIdx.x * 2;     // first of two adjacent N-tiles

  const int arow_idx = wave * 16 + m;  // batch row this lane loads for A
  const _Float16* __restrict__ wrow0 = Wt + (size_t)(nt0 * 16 + m) * Ktot;
  const _Float16* __restrict__ wrow1 = wrow0 + (size_t)16 * Ktot;

  v8f acc00 = {}, acc01 = {};          // tile0: even/odd K-block chains
  v8f acc10 = {}, acc11 = {};          // tile1: even/odd K-block chains

  // Segment 0: feed-forward input Ax (K = 0..K0), K0 % 64 == 0
  {
    const _Float16* __restrict__ ar = Ax + (size_t)arow_idx * sAx;
#pragma unroll 2
    for (int kb = 0; kb < K0; kb += 64) {
      v16h a0 = load_a_frag(ar, kb, hh);
      v16h a1 = load_a_frag(ar, kb + 32, hh);
      v16h b00 = load_b_frag(wrow0, kb, hh);
      v16h b10 = load_b_frag(wrow1, kb, hh);
      v16h b01 = load_b_frag(wrow0, kb + 32, hh);
      v16h b11 = load_b_frag(wrow1, kb + 32, hh);
      acc00 = WMMA_F16(a0, b00, acc00);
      acc10 = WMMA_F16(a0, b10, acc10);
      acc01 = WMMA_F16(a1, b01, acc01);
      acc11 = WMMA_F16(a1, b11, acc11);
    }
  }
  // Segment 1: recurrent spikes As (K = K0..Ktot), K1 % 64 == 0
  {
    const _Float16* __restrict__ ar = As + (size_t)arow_idx * K1;
    const _Float16* __restrict__ w0 = wrow0 + K0;
    const _Float16* __restrict__ w1 = wrow1 + K0;
#pragma unroll 2
    for (int kb = 0; kb < K1; kb += 64) {
      v16h a0 = load_a_frag(ar, kb, hh);
      v16h a1 = load_a_frag(ar, kb + 32, hh);
      v16h b00 = load_b_frag(w0, kb, hh);
      v16h b10 = load_b_frag(w1, kb, hh);
      v16h b01 = load_b_frag(w0, kb + 32, hh);
      v16h b11 = load_b_frag(w1, kb + 32, hh);
      acc00 = WMMA_F16(a0, b00, acc00);
      acc10 = WMMA_F16(a0, b10, acc10);
      acc01 = WMMA_F16(a1, b01, acc01);
      acc11 = WMMA_F16(a1, b11, acc11);
    }
  }

  v8f accT[2];
  accT[0] = acc00 + acc01;
  accT[1] = acc10 + acc11;

  // Epilogue: LIF state update. C/D layout: lane n = L%16; VGPR r -> row 8*hh + r.
#pragma unroll
  for (int tile = 0; tile < 2; ++tile) {
    const int gcol = (nt0 + tile) * 16 + m;
    const float bn = bias[gcol];
#pragma unroll
    for (int r = 0; r < 8; ++r) {
      int grow = wave * 16 + 8 * hh + r;
      size_t idx = (size_t)grow * N + gcol;
      float cn = ALPHA * c[idx] + accT[tile][r] + bn;
      float vn = BETA * v[idx] + cn;
      float sp = (vn - THETA) > 0.0f ? 1.0f : 0.0f;
      vn -= sp * THETA;
      c[idx] = cn;
      v[idx] = vn;
      s_out[idx] = (_Float16)sp;
      if (y_out) y_out[idx] = sp;
    }
  }
}

// ---------------------------------------------------------------------------
// Host launcher
// ---------------------------------------------------------------------------
extern "C" void kernel_launch(void* const* d_in, const int* in_sizes, int n_in,
                              void* d_out, int out_size, void* d_ws, size_t ws_size,
                              hipStream_t stream) {
  (void)in_sizes; (void)n_in; (void)out_size; (void)ws_size;

  const float* x     = (const float*)d_in[0];  // (B,T,D)
  const float* Win1  = (const float*)d_in[1];  // (D,H1)
  const float* Wrec1 = (const float*)d_in[2];  // (H1,H1)
  const float* b1    = (const float*)d_in[3];  // (H1)
  const float* Win2  = (const float*)d_in[4];  // (H1,H2)
  const float* Wrec2 = (const float*)d_in[5];  // (H2,H2)
  const float* b2    = (const float*)d_in[6];  // (H2)
  float* out = (float*)d_out;                  // (B,H2)

  const int K1tot = DDIM + H1;  // 1536
  const int K2tot = H1 + H2;    // 2048

  // Workspace layout (256B aligned sections)
  char* ws = (char*)d_ws;
  size_t off = 0;
  auto take = [&](size_t bytes) {
    size_t o = off;
    off = (off + bytes + 255) & ~(size_t)255;
    return o;
  };
  size_t xh_off  = take((size_t)BDIM * TDIM * DDIM * 2);
  size_t w1t_off = take((size_t)H1 * K1tot * 2);
  size_t w2t_off = take((size_t)H2 * K2tot * 2);
  size_t state_begin = off;
  size_t c1_off  = take((size_t)BDIM * H1 * 4);
  size_t v1_off  = take((size_t)BDIM * H1 * 4);
  size_t c2_off  = take((size_t)BDIM * H2 * 4);
  size_t v2_off  = take((size_t)BDIM * H2 * 4);
  size_t s1a_off = take((size_t)BDIM * H1 * 2);
  size_t s1b_off = take((size_t)BDIM * H1 * 2);
  size_t s2a_off = take((size_t)BDIM * H2 * 2);
  size_t s2b_off = take((size_t)BDIM * H2 * 2);
  size_t state_end = off;

  _Float16* Xh  = (_Float16*)(ws + xh_off);
  _Float16* W1t = (_Float16*)(ws + w1t_off);
  _Float16* W2t = (_Float16*)(ws + w2t_off);
  float* c1 = (float*)(ws + c1_off);
  float* v1 = (float*)(ws + v1_off);
  float* c2 = (float*)(ws + c2_off);
  float* v2 = (float*)(ws + v2_off);
  _Float16* s1[2] = { (_Float16*)(ws + s1a_off), (_Float16*)(ws + s1b_off) };
  _Float16* s2[2] = { (_Float16*)(ws + s2a_off), (_Float16*)(ws + s2b_off) };

  // Prep: convert x to f16; pack transposed f16 weights; zero all state.
  {
    int n = BDIM * TDIM * DDIM;
    f32_to_f16_kernel<<<(n + 255) / 256, 256, 0, stream>>>(x, Xh, n);
  }
  {
    int n = H1 * K1tot;
    pack_wt_kernel<<<(n + 255) / 256, 256, 0, stream>>>(Win1, Wrec1, W1t, DDIM, H1, H1);
  }
  {
    int n = H2 * K2tot;
    pack_wt_kernel<<<(n + 255) / 256, 256, 0, stream>>>(Win2, Wrec2, W2t, H1, H2, H2);
  }
  hipMemsetAsync(ws + state_begin, 0, state_end - state_begin, stream);

  // Sequential scan over T: layer1 step then layer2 step, double-buffered spikes.
  for (int t = 0; t < TDIM; ++t) {
    int prev = t & 1, cur = prev ^ 1;
    // Layer 1: A = [x_t | s1_prev]
    lif_step_kernel<<<H1 / 32, 128, 0, stream>>>(
        Xh + (size_t)t * DDIM, TDIM * DDIM, DDIM,
        s1[prev], H1, W1t, b1, c1, v1, s1[cur], nullptr, H1);
    // Layer 2: A = [s1_cur | s2_prev]
    float* y = (t == TDIM - 1) ? out : nullptr;
    lif_step_kernel<<<H2 / 32, 128, 0, stream>>>(
        s1[cur], H1, H1,
        s2[prev], H2, W2t, b2, c2, v2, s2[cur], y, H2);
  }
}
